// GNN_SAGE_20993800143187
// MI455X (gfx1250) — matrix operand order repaired and checked by reading, not compiled, not run on hardware
//
#include <hip/hip_runtime.h>

#define IN_DIM 23
#define HID_DIM 128
#define OUT_DIM 64
#define NUM_GRAPHS 64

typedef __attribute__((ext_vector_type(16))) __bf16 v16bf;
typedef __attribute__((ext_vector_type(16))) unsigned short v16u16;
typedef __attribute__((ext_vector_type(8))) float v8f;

union FragU {
    v16u16 v;
    uint4  q[2];
};

// ---------------------------------------------------------------- zero scratch
__global__ void zero_kernel(float* p, long long n) {
    long long i = (long long)blockIdx.x * blockDim.x + threadIdx.x;
    long long stride = (long long)gridDim.x * blockDim.x;
    for (; i < n; i += stride) p[i] = 0.0f;
}

// ------------------------------------------- edge pass 1: degree + 23-dim agg
// one wave per edge; lanes 0..22 carry feature dims
__global__ __launch_bounds__(256) void edge_agg1_kernel(
    const int* __restrict__ src, const int* __restrict__ dst,
    const float* __restrict__ feat, float* __restrict__ deg,
    float* __restrict__ agg1, int nEdges)
{
    int e = blockIdx.x * 8 + (threadIdx.x >> 5);
    int lane = threadIdx.x & 31;
    if (e >= nEdges) return;
    int s = src[e], d = dst[e];
    if (lane == 0) atomicAdd(&deg[d], 1.0f);
    if (lane < IN_DIM)
        atomicAdd(&agg1[(long long)d * IN_DIM + lane],
                  feat[(long long)s * IN_DIM + lane]);
}

// ------------------------- layer 1: (agg+h)/(deg+1) @ W1 + b1, relu, L2 norm
// block = 256 threads = 8 waves; 16 nodes x 128 outputs per block
__global__ __launch_bounds__(256) void layer1_kernel(
    const float* __restrict__ feat, const float* __restrict__ agg1,
    const float* __restrict__ deg, const float* __restrict__ W1,
    const float* __restrict__ b1, float* __restrict__ h1, int nNodes)
{
    __shared__ __align__(16) __bf16 sA[16 * 32];  // h_neigh tile, K padded to 32
    __shared__ __align__(16) float  sH[16 * HID_DIM];
    __shared__ float sP[256];
    __shared__ float sInv[16];

    int m0 = blockIdx.x * 16;
    int tid = threadIdx.x;

    // build 16x32 bf16 A tile (K 23..31 zero-padded)
    for (int idx = tid; idx < 16 * 32; idx += 256) {
        int m = idx >> 5;
        int k = idx & 31;
        int node = m0 + m;
        float v = 0.0f;
        if (node < nNodes && k < IN_DIM) {
            float dd = deg[node] + 1.0f;
            long long off = (long long)node * IN_DIM + k;
            v = (agg1[off] + feat[off]) / dd;
        }
        sA[idx] = (__bf16)v;
    }
    __syncthreads();

    int lane = tid & 31;
    int wave = tid >> 5;        // 0..7 -> N tile
    int half = lane >> 4;
    int l15  = lane & 15;
    int n    = wave * 16 + l15; // output column 0..127

    // A fragment: two contiguous 16B chunks per lane (16-bit 16x32 layout)
    const uint4* sA4 = (const uint4*)sA;       // 8 bf16 per uint4, 4 per row
    FragU ua;
    ua.q[0] = sA4[l15 * 4 + half];             // K = half*8 .. half*8+7
    ua.q[1] = sA4[l15 * 4 + 2 + half];         // K = 16+half*8 .. +7
    v16bf a = __builtin_bit_cast(v16bf, ua.v);

    // B fragment: W1 is [23][128], K-major
    v16bf b;
    #pragma unroll
    for (int i = 0; i < 16; ++i) {
        int K = half * 8 + i + (i >= 8 ? 8 : 0);
        float w = (K < IN_DIM) ? W1[K * HID_DIM + n] : 0.0f;
        b[i] = (__bf16)w;
    }

    v8f c = {};
    c = __builtin_amdgcn_wmma_f32_16x16x32_bf16(false, a, false, b,
                                                (short)0, c, false, false);
    // bias + relu -> LDS tile
    float bias = b1[n];
    #pragma unroll
    for (int j = 0; j < 8; ++j) {
        int M = j + half * 8;
        float v = c[j] + bias;
        sH[M * HID_DIM + n] = v > 0.0f ? v : 0.0f;
    }
    __syncthreads();

    // parallel row L2-norm: 16 threads/row, 8-element partials
    {
        int m = tid >> 4;
        int part = tid & 15;
        const float* row = &sH[m * HID_DIM + part * 8];
        float s = 0.0f;
        #pragma unroll
        for (int j = 0; j < 8; ++j) s += row[j] * row[j];
        sP[tid] = s;
    }
    __syncthreads();
    if (tid < 16) {
        float s = 0.0f;
        #pragma unroll
        for (int p = 0; p < 16; ++p) s += sP[tid * 16 + p];
        sInv[tid] = 1.0f / fmaxf(sqrtf(s), 1e-12f);
    }
    __syncthreads();

    // write normalized h1 as float4 (coalesced b128 stores)
    const float4* sH4 = (const float4*)sH;
    float4* h1_4 = (float4*)h1;
    for (int idx = tid; idx < 16 * (HID_DIM / 4); idx += 256) {
        int m = idx >> 5;                 // 32 float4 per row
        int node = m0 + m;
        if (node < nNodes) {
            float4 v = sH4[idx];
            float s = sInv[m];
            v.x *= s; v.y *= s; v.z *= s; v.w *= s;
            h1_4[(long long)node * (HID_DIM / 4) + (idx & 31)] = v;
        }
    }
}

// ---------------------------------------- edge pass 2: 128-dim aggregation
// one wave per edge; 4 coalesced 32-lane atomic bursts per edge
__global__ __launch_bounds__(256) void edge_agg2_kernel(
    const int* __restrict__ src, const int* __restrict__ dst,
    const float* __restrict__ h1, float* __restrict__ agg2, int nEdges)
{
    int e = blockIdx.x * 8 + (threadIdx.x >> 5);
    int lane = threadIdx.x & 31;
    if (e >= nEdges) return;
    long long s = (long long)src[e] * HID_DIM;
    long long d = (long long)dst[e] * HID_DIM;
    #pragma unroll
    for (int i = 0; i < 4; ++i) {
        int k = lane + 32 * i;
        atomicAdd(&agg2[d + k], h1[s + k]);
    }
}

// --------------- layer 2: (agg2+h1)/(deg+1) @ W2 + b2, per-graph atomic pool
// block = 128 threads = 4 waves; 16 nodes x 64 outputs; K=128 -> 4 WMMA steps
__global__ __launch_bounds__(128) void layer2_kernel(
    const float* __restrict__ h1, const float* __restrict__ agg2,
    const float* __restrict__ deg, const float* __restrict__ W2,
    const float* __restrict__ b2, const int* __restrict__ graph_ids,
    float* __restrict__ gsum, int nNodes)
{
    __shared__ __align__(16) __bf16 sA[16 * HID_DIM];  // h_neigh2 tile

    int m0 = blockIdx.x * 16;
    int tid = threadIdx.x;

    // build tile with float4 loads: 16 rows x 32 quads
    const float4* h1_4   = (const float4*)h1;
    const float4* agg2_4 = (const float4*)agg2;
    for (int idx = tid; idx < 16 * (HID_DIM / 4); idx += 128) {
        int m  = idx >> 5;
        int kq = idx & 31;
        int node = m0 + m;
        float4 v = {0.0f, 0.0f, 0.0f, 0.0f};
        if (node < nNodes) {
            float inv = 1.0f / (deg[node] + 1.0f);
            long long off = (long long)node * (HID_DIM / 4) + kq;
            float4 av = agg2_4[off];
            float4 hv = h1_4[off];
            v.x = (av.x + hv.x) * inv;
            v.y = (av.y + hv.y) * inv;
            v.z = (av.z + hv.z) * inv;
            v.w = (av.w + hv.w) * inv;
        }
        __bf16* o = &sA[m * HID_DIM + kq * 4];
        o[0] = (__bf16)v.x; o[1] = (__bf16)v.y;
        o[2] = (__bf16)v.z; o[3] = (__bf16)v.w;
    }
    __syncthreads();

    int lane = tid & 31;
    int wave = tid >> 5;        // 0..3
    int half = lane >> 4;
    int l15  = lane & 15;
    int n    = wave * 16 + l15; // output column 0..63

    const uint4* sA4 = (const uint4*)sA;   // 16 uint4 per row
    v8f c = {};
    #pragma unroll
    for (int kk = 0; kk < 4; ++kk) {
        FragU ua;
        ua.q[0] = sA4[l15 * 16 + kk * 4 + half];
        ua.q[1] = sA4[l15 * 16 + kk * 4 + 2 + half];
        v16bf a = __builtin_bit_cast(v16bf, ua.v);

        v16bf b;
        #pragma unroll
        for (int i = 0; i < 16; ++i) {
            int Kb = kk * 32 + half * 8 + i + (i >= 8 ? 8 : 0);
            b[i] = (__bf16)W2[Kb * OUT_DIM + n];
        }
        c = __builtin_amdgcn_wmma_f32_16x16x32_bf16(false, a, false, b,
                                                    (short)0, c, false, false);
    }

    float bias = b2[n];
    #pragma unroll
    for (int j = 0; j < 8; ++j) {
        int M = j + half * 8;
        int node = m0 + M;
        if (node < nNodes) {
            int g = graph_ids[node];
            atomicAdd(&gsum[g * OUT_DIM + n], c[j] + bias);
        }
    }
}

// ----------------------------------------------------------- per-graph counts
__global__ __launch_bounds__(256) void count_kernel(
    const int* __restrict__ graph_ids, float* __restrict__ cnt, int nNodes)
{
    int i = blockIdx.x * blockDim.x + threadIdx.x;
    if (i < nNodes) atomicAdd(&cnt[graph_ids[i]], 1.0f);
}

// --------------------------------------------------------------- finalize out
__global__ __launch_bounds__(256) void finalize_kernel(
    const float* __restrict__ gsum, const float* __restrict__ cnt,
    float* __restrict__ out)
{
    int i = blockIdx.x * blockDim.x + threadIdx.x;
    if (i < NUM_GRAPHS * OUT_DIM) {
        int g = i / OUT_DIM;
        out[i] = gsum[i] / fmaxf(cnt[g], 1.0f);
    }
}

extern "C" void kernel_launch(void* const* d_in, const int* in_sizes, int n_in,
                              void* d_out, int out_size, void* d_ws, size_t ws_size,
                              hipStream_t stream) {
    const float* feat = (const float*)d_in[0];
    const float* W1   = (const float*)d_in[1];
    const float* b1   = (const float*)d_in[2];
    const float* W2   = (const float*)d_in[3];
    const float* b2   = (const float*)d_in[4];
    const int*   src  = (const int*)d_in[5];
    const int*   dst  = (const int*)d_in[6];
    const int*   gid  = (const int*)d_in[7];

    int nEdges = in_sizes[5];
    int nNodes = in_sizes[7];

    // workspace layout: [deg | agg1 | agg2 | gsum | cnt | h1]
    float* ws   = (float*)d_ws;
    float* deg  = ws;
    float* agg1 = deg  + nNodes;
    float* agg2 = agg1 + (long long)nNodes * IN_DIM;
    float* gsum = agg2 + (long long)nNodes * HID_DIM;
    float* cnt  = gsum + NUM_GRAPHS * OUT_DIM;
    float* h1   = cnt  + NUM_GRAPHS;

    long long zeroTotal = (long long)nNodes * (1 + IN_DIM + HID_DIM)
                        + NUM_GRAPHS * OUT_DIM + NUM_GRAPHS;

    zero_kernel<<<4096, 256, 0, stream>>>(ws, zeroTotal);

    int edgeBlocks = (nEdges + 7) / 8;
    edge_agg1_kernel<<<edgeBlocks, 256, 0, stream>>>(src, dst, feat, deg, agg1, nEdges);

    int nodeTiles = (nNodes + 15) / 16;
    layer1_kernel<<<nodeTiles, 256, 0, stream>>>(feat, agg1, deg, W1, b1, h1, nNodes);

    edge_agg2_kernel<<<edgeBlocks, 256, 0, stream>>>(src, dst, h1, agg2, nEdges);

    layer2_kernel<<<nodeTiles, 128, 0, stream>>>(h1, agg2, deg, W2, b2, gid, gsum, nNodes);

    count_kernel<<<(nNodes + 255) / 256, 256, 0, stream>>>(gid, cnt, nNodes);

    finalize_kernel<<<(NUM_GRAPHS * OUT_DIM + 255) / 256, 256, 0, stream>>>(
        gsum, cnt, (float*)d_out);
}
